// DSI_beta_82059645157486
// MI455X (gfx1250) — compile-verified, matrix operand order checked by reading, not run on hardware
//
#include <hip/hip_runtime.h>
#include <hip/hip_bf16.h>

// Problem constants (match reference setup_inputs).
#define BATCH 256
#define SEQ   512
#define DM    512
#define NROWS (BATCH * SEQ)        // 131072
#define NHOP  3
#define TEMP_ 10.0f
#define SLOPE_ 0.01f
#define EPS_COS_ 1e-8f
#define EPS_LN_  1e-5f

// LDS weight slab: 128 cols x 512 K f16, row stride padded to 520 halves
// (1040 B = 65*16 B: keeps 16B alignment for b128 loads and spreads the 16
// lanes across LDS banks 0,4,...,60 instead of all hitting one bank).
#define LDSTRIDE 520
#define GEMM_SMEM_BYTES (512 + 128 * LDSTRIDE * 2)   // distS + ldsW = 133,632 B

typedef __attribute__((ext_vector_type(16))) _Float16 v16h;
typedef __attribute__((ext_vector_type(8)))  _Float16 v8h;
typedef __attribute__((ext_vector_type(8)))  float    v8f;
typedef __attribute__((ext_vector_type(4)))  unsigned int u32x4;
typedef __attribute__((ext_vector_type(8)))  int i32x8;
typedef __attribute__((ext_vector_type(4)))  int i32x4;

#if defined(__has_builtin)
#if __has_builtin(__builtin_amdgcn_tensor_load_to_lds)
#define HAVE_TDM 1
#endif
#endif

// ---------------------------------------------------------------- helpers
__device__ __forceinline__ float waveReduceSum(float v) {
#pragma unroll
    for (int o = 16; o > 0; o >>= 1) v += __shfl_xor(v, o, 32);
    return v;
}
__device__ __forceinline__ float waveReduceMax(float v) {
#pragma unroll
    for (int o = 16; o > 0; o >>= 1) v = fmaxf(v, __shfl_xor(v, o, 32));
    return v;
}
// Block of 512 threads (16 waves): every thread returns the block total.
__device__ __forceinline__ float blockReduceSum512(float v, float* red, int wave, int lane) {
    v = waveReduceSum(v);
    __syncthreads();                 // protect red[] against previous use
    if (lane == 0) red[wave] = v;
    __syncthreads();
    float t = red[lane & 15];
#pragma unroll
    for (int o = 8; o > 0; o >>= 1) t += __shfl_xor(t, o, 32);
    return t;
}

// CDNA5 16-bit A-matrix 16x32 fragment (ISA 7.12.2):
// lane L holds row M = L%16; lane-group g = L/16.
// halves 0..7  <- K = g*8 + (0..7);  halves 8..15 <- K = 16 + g*8 + (0..7)
__device__ __forceinline__ v16h load_a_frag(const float* __restrict__ p, int g) {
    const float4* p0 = (const float4*)(p + g * 8);
    const float4* p1 = (const float4*)(p + 16 + g * 8);
    float4 x0 = p0[0], x1 = p0[1], y0 = p1[0], y1 = p1[1];
    v16h r;
    r[0]  = (_Float16)x0.x; r[1]  = (_Float16)x0.y; r[2]  = (_Float16)x0.z; r[3]  = (_Float16)x0.w;
    r[4]  = (_Float16)x1.x; r[5]  = (_Float16)x1.y; r[6]  = (_Float16)x1.z; r[7]  = (_Float16)x1.w;
    r[8]  = (_Float16)y0.x; r[9]  = (_Float16)y0.y; r[10] = (_Float16)y0.z; r[11] = (_Float16)y0.w;
    r[12] = (_Float16)y1.x; r[13] = (_Float16)y1.y; r[14] = (_Float16)y1.z; r[15] = (_Float16)y1.w;
    return r;
}
// f16-source fragment: two 16B loads, no conversion VALU (global or LDS).
__device__ __forceinline__ v16h load_a_frag(const _Float16* __restrict__ p, int g) {
    v8h lo = *(const v8h*)(p + g * 8);
    v8h hi = *(const v8h*)(p + 16 + g * 8);
    v16h r;
#pragma unroll
    for (int i = 0; i < 8; ++i) { r[i] = lo[i]; r[8 + i] = hi[i]; }
    return r;
}
__device__ __forceinline__ v8f wmma_f16(v16h a, v16h b, v8f c) {
    return __builtin_amdgcn_wmma_f32_16x16x32_f16(
        /*neg_a=*/false, a, /*neg_b=*/false, b,
        /*c_mod=*/(short)0, c, /*reuse_a=*/false, /*reuse_b=*/false);
}

// 16 consecutive elements -> float regs (overloads for fp32 / f16 rows).
__device__ __forceinline__ void load16f(const float* __restrict__ p, float* o) {
#pragma unroll
    for (int i = 0; i < 16; i += 4) {
        float4 v = *(const float4*)(p + i);
        o[i] = v.x; o[i + 1] = v.y; o[i + 2] = v.z; o[i + 3] = v.w;
    }
}
__device__ __forceinline__ void load16f(const _Float16* __restrict__ p, float* o) {
    v8h a = ((const v8h*)p)[0], b = ((const v8h*)p)[1];
#pragma unroll
    for (int i = 0; i < 8; ++i) { o[i] = (float)a[i]; o[8 + i] = (float)b[i]; }
}

// ---------------------------------------------------------------- kernels
// W fp32 [3,512,512] -> f16 same layout (row e major, contiguous d).
__global__ __launch_bounds__(256) void convW_kernel(const float* __restrict__ w,
                                                    _Float16* __restrict__ wh, int n) {
    int i = blockIdx.x * 256 + threadIdx.x;
    if (i < n) wh[i] = (_Float16)w[i];
}

// qn[b] = ||q[b,:]||  (one wave per batch row)
__global__ __launch_bounds__(256) void qnorm_kernel(const float* __restrict__ q,
                                                    float* __restrict__ qn) {
    const int b = blockIdx.x * 8 + (threadIdx.x >> 5);
    const int lane = threadIdx.x & 31;
    const float* p = q + (size_t)b * DM + lane * 16;
    float v[16], ss = 0.f;
    load16f(p, v);
#pragma unroll
    for (int i = 0; i < 16; ++i) ss += v[i] * v[i];
    ss = waveReduceSum(ss);
    if (lane == 0) qn[b] = sqrtf(ss);
}

// logits[b,s] = TEMP * (q.s_row) / max(|q||s_row|, eps). One wave per row;
// computes the row norm in the same single pass over s.
template <typename T>
__global__ __launch_bounds__(256) void dot_cos_kernel(const float* __restrict__ q,
                                                      const T* __restrict__ s,
                                                      const float* __restrict__ qn,
                                                      float* __restrict__ logits) {
    const int row  = blockIdx.x * 8 + (threadIdx.x >> 5);
    const int lane = threadIdx.x & 31;
    const int b = row >> 9;
    float sv[16], qv[16];
    load16f(s + (size_t)row * DM + lane * 16, sv);
    load16f(q + (size_t)b * DM + lane * 16, qv);
    float dot = 0.f, ss = 0.f;
#pragma unroll
    for (int i = 0; i < 16; ++i) { dot += sv[i] * qv[i]; ss += sv[i] * sv[i]; }
    dot = waveReduceSum(dot);
    ss  = waveReduceSum(ss);
    if (lane == 0) {
        float denom = fmaxf(qn[b] * sqrtf(ss), EPS_COS_);
        logits[row] = TEMP_ * dot / denom;
    }
}

// softmax over S=512 per batch; one block of 512 threads per batch (in-place safe).
__global__ __launch_bounds__(512) void softmax512_kernel(const float* __restrict__ in,
                                                         float* __restrict__ out) {
    __shared__ float red[16];
    const int b = blockIdx.x, t = threadIdx.x;
    const int wave = t >> 5, lane = t & 31;
    float v = in[(size_t)b * SEQ + t];
    float m = waveReduceMax(v);
    if (lane == 0) red[wave] = m;
    __syncthreads();
    float mAll = red[lane & 15];
#pragma unroll
    for (int o = 8; o > 0; o >>= 1) mAll = fmaxf(mAll, __shfl_xor(mAll, o, 32));
    float e = __expf(v - mAll);
    float s = waveReduceSum(e);
    __syncthreads();                 // all reads of red (max phase) done
    if (lane == 0) red[wave] = s;
    __syncthreads();
    float tot = red[lane & 15];
#pragma unroll
    for (int o = 8; o > 0; o >>= 1) tot += __shfl_xor(tot, o, 32);
    out[(size_t)b * SEQ + t] = e / tot;
}

__global__ __launch_bounds__(256) void fillzero_kernel(float* __restrict__ p, int n) {
    int i = blockIdx.x * 256 + threadIdx.x;
    if (i < n) p[i] = 0.f;
}

// Core WMMA GEMM: sOut[row,e] = f16(leaky_relu(sum_d sIn[row,d]*W[e,d] + bias[e])),
// fused epilogue accumulates agg[b,e] += dist[row]*x (fp32).
// Block tile 128 rows x 128 cols; 8 waves, each 32x64 (2x4 frags, 8 WMMA/K-step).
// The block's 128x512 f16 weight slab is staged into LDS by the Tensor Data
// Mover (one descriptor, TENSORcnt-tracked, LDS row padding for bank spread).
template <typename AT>
__global__ __launch_bounds__(256) void gemm_hop_kernel(const AT* __restrict__ sIn,
                                                       const _Float16* __restrict__ Wh,
                                                       const float* __restrict__ bias,
                                                       const float* __restrict__ dist,
                                                       _Float16* __restrict__ sOut,
                                                       float* __restrict__ agg) {
    extern __shared__ char smemRaw[];
    float*    distS = (float*)smemRaw;            // 128 floats
    _Float16* ldsW  = (_Float16*)(smemRaw + 512); // [128][LDSTRIDE] f16

    const int tid  = threadIdx.x;
    const int wave = tid >> 5, lane = tid & 31;
    const int g = lane >> 4, ln = lane & 15;
    const int wm = wave >> 1, wn = wave & 1;
    const int rowBase = blockIdx.y << 7;                  // 128-row block tile
    const int colWave = (blockIdx.x << 7) + (wn << 6);    // wave's 64 cols (global)
    const int rowWave = rowBase + (wm << 5);              // wave's 32 rows
    const _Float16* WhBlk = Wh + (size_t)(blockIdx.x << 7) * DM;

#if defined(HAVE_TDM)
    if (wave == 0) {
        // Tensor DMA descriptor: 2D tile 512(K,f16) x 128(rows), global->LDS,
        // pad 4 DWORDs after every 256 DWORDs (one 1024B row) -> 1040B LDS rows.
        unsigned long long gaddr = (unsigned long long)(uintptr_t)WhBlk;
        u32x4 g0;
        g0[0] = 1u;                                          // count=1 (user D#)
        g0[1] = (unsigned int)(uintptr_t)ldsW;               // LDS byte address
        g0[2] = (unsigned int)gaddr;                         // global_addr[31:0]
        g0[3] = (unsigned int)((gaddr >> 32) & 0x01FFFFFFu)  // global_addr[56:32]
                | (2u << 30);                                // type=2 (image)
        i32x8 g1;
        g1[0] = (int)0x07D10000u;   // data_size=2B, pad_enable, interval=256dw, amount=4dw
        g1[1] = (int)(512u << 16);  // tensor_dim0 = 512
        g1[2] = (int)(128u << 16);  // tensor_dim1 = 128
        g1[3] = (int)(512u << 16);  // tile_dim0 = 512
        g1[4] = 128;                // tile_dim1 = 128
        g1[5] = 512;                // tensor_dim0_stride = 512
        g1[6] = 0; g1[7] = 0;
        i32x4 z4 = {0, 0, 0, 0};
#if __clang_major__ >= 23
        i32x8 z8 = {0, 0, 0, 0, 0, 0, 0, 0};
        __builtin_amdgcn_tensor_load_to_lds(g0, g1, z4, z4, z8, 0);
#else
        __builtin_amdgcn_tensor_load_to_lds(g0, g1, z4, z4, 0);
#endif
#if __has_builtin(__builtin_amdgcn_s_wait_tensorcnt)
        __builtin_amdgcn_s_wait_tensorcnt(0);
#else
        asm volatile("s_wait_tensorcnt 0x0" ::: "memory");
#endif
    }
#else
    // Fallback: cooperative global->LDS copy with the same padded layout.
    for (int i = tid; i < 128 * (DM / 8); i += 256) {
        int col = i / (DM / 8), seg = i % (DM / 8);
        ((v8h*)(ldsW + col * LDSTRIDE))[seg] = ((const v8h*)(WhBlk + col * DM))[seg];
    }
#endif
    if (tid < 128) distS[tid] = dist[rowBase + tid];
    __syncthreads();

    const AT* a0p = sIn + (size_t)(rowWave + ln) * DM;        // M 0..15
    const AT* a1p = sIn + (size_t)(rowWave + 16 + ln) * DM;   // M 16..31
    const _Float16* bb = ldsW + ((wn << 6) + ln) * LDSTRIDE;  // wave's LDS W slab

    v8f acc[2][4] = {};
    for (int k = 0; k < DM; k += 32) {
        __builtin_prefetch(a0p + k + 256, 0, 1);   // global_prefetch_b8 on streamed A
        __builtin_prefetch(a1p + k + 256, 0, 1);
        v16h a0 = load_a_frag(a0p + k, g);
        v16h a1 = load_a_frag(a1p + k, g);
        v16h b0 = load_a_frag(bb + k, g);          // ds_load from staged slab
        v16h b1 = load_a_frag(bb + 16 * LDSTRIDE + k, g);
        v16h b2 = load_a_frag(bb + 32 * LDSTRIDE + k, g);
        v16h b3 = load_a_frag(bb + 48 * LDSTRIDE + k, g);
        acc[0][0] = wmma_f16(a0, b0, acc[0][0]);
        acc[0][1] = wmma_f16(a0, b1, acc[0][1]);
        acc[0][2] = wmma_f16(a0, b2, acc[0][2]);
        acc[0][3] = wmma_f16(a0, b3, acc[0][3]);
        acc[1][0] = wmma_f16(a1, b0, acc[1][0]);
        acc[1][1] = wmma_f16(a1, b1, acc[1][1]);
        acc[1][2] = wmma_f16(a1, b2, acc[1][2]);
        acc[1][3] = wmma_f16(a1, b3, acc[1][3]);
    }

    // Epilogue. C/D layout: VGPR r, lanes 0-15 -> M=r, lanes 16-31 -> M=8+r; N=lane%16.
    const int b = rowBase >> 9;
    float biasv[4], aggP[4] = {0.f, 0.f, 0.f, 0.f};
#pragma unroll
    for (int ni = 0; ni < 4; ++ni) biasv[ni] = bias[colWave + ni * 16 + ln];
#pragma unroll
    for (int mi = 0; mi < 2; ++mi) {
#pragma unroll
        for (int r = 0; r < 8; ++r) {
            const int rt = (wm << 5) + (mi << 4) + (g << 3) + r;   // row within block tile
            const float dw = distS[rt];
            const size_t rowOff = (size_t)(rowBase + rt) * DM + colWave;
#pragma unroll
            for (int ni = 0; ni < 4; ++ni) {
                float x = acc[mi][ni][r] + biasv[ni];
                x = x > 0.f ? x : SLOPE_ * x;                       // LeakyReLU(0.01)
                sOut[rowOff + ni * 16 + ln] = (_Float16)x;
                aggP[ni] += dw * x;
            }
        }
    }
#pragma unroll
    for (int ni = 0; ni < 4; ++ni) aggP[ni] += __shfl_xor(aggP[ni], 16, 32);
    if (lane < 16) {
#pragma unroll
        for (int ni = 0; ni < 4; ++ni)
            atomicAdd(agg + (size_t)b * DM + colWave + ni * 16 + ln, aggP[ni]);
    }
}

// q_out = LayerNorm(q_in + agg); also emits ||q_out|| for the next cosine pass.
__global__ __launch_bounds__(512) void ln512_kernel(const float* __restrict__ qIn,
                                                    const float* __restrict__ agg,
                                                    const float* __restrict__ gamma,
                                                    const float* __restrict__ beta,
                                                    float* __restrict__ qOut,
                                                    float* __restrict__ qn) {
    __shared__ float red[16];
    const int b = blockIdx.x, t = threadIdx.x;
    const int wave = t >> 5, lane = t & 31;
    float x = qIn[(size_t)b * DM + t] + agg[(size_t)b * DM + t];
    float mu = blockReduceSum512(x, red, wave, lane) * (1.0f / DM);
    float d = x - mu;
    float var = blockReduceSum512(d * d, red, wave, lane) * (1.0f / DM);
    float y = d * rsqrtf(var + EPS_LN_) * gamma[t] + beta[t];
    qOut[(size_t)b * DM + t] = y;
    float nrm = blockReduceSum512(y * y, red, wave, lane);
    if (t == 0) qn[b] = sqrtf(nrm);
}

// ---------------------------------------------------------------- launch
extern "C" void kernel_launch(void* const* d_in, const int* in_sizes, int n_in,
                              void* d_out, int out_size, void* d_ws, size_t ws_size,
                              hipStream_t stream) {
    (void)in_sizes; (void)n_in; (void)out_size; (void)ws_size;
    const float* query = (const float*)d_in[0];   // [256,512]
    const float* state = (const float*)d_in[1];   // [256,512,512]
    const float* W     = (const float*)d_in[2];   // [3,512,512]
    const float* bias  = (const float*)d_in[3];   // [3,512]
    const float* gamma = (const float*)d_in[4];   // [512]
    const float* beta  = (const float*)d_in[5];   // [512]

    float* outQ    = (float*)d_out;               // [256,512]
    float* outDist = outQ + (size_t)BATCH * DM;   // [256,512]

    // Workspace (needs ~260 MB): two f16 ping-pong s buffers + small tensors.
    const size_t SELEMS = (size_t)NROWS * DM;     // 67,108,864 elements
    _Float16* s1   = (_Float16*)d_ws;             // 128 MB
    _Float16* s2   = s1 + SELEMS;                 // 128 MB
    _Float16* Wh   = s2 + SELEMS;                 // [3,512,512] f16
    float*    dist = (float*)(Wh + (size_t)NHOP * DM * DM);
    float*    agg  = dist + NROWS;                // [256,512]
    float*    qA   = agg + (size_t)BATCH * DM;
    float*    qB   = qA + (size_t)BATCH * DM;
    float*    qn   = qB + (size_t)BATCH * DM;     // [256]

    // One-time W fp32 -> f16 (1.5 MB, L2-resident across all GEMMs).
    convW_kernel<<<(NHOP * DM * DM) / 256, 256, 0, stream>>>(W, Wh, NHOP * DM * DM);
    qnorm_kernel<<<BATCH / 8, 256, 0, stream>>>(query, qn);

    const float* qCur = query;
    _Float16* sNext = s1;
    _Float16* sPrev = nullptr;                    // f16 s after hop>=1
    for (int hop = 0; hop < NHOP; ++hop) {
        if (hop == 0)
            dot_cos_kernel<float><<<NROWS / 8, 256, 0, stream>>>(qCur, state, qn, dist);
        else
            dot_cos_kernel<_Float16><<<NROWS / 8, 256, 0, stream>>>(qCur, sPrev, qn, dist);
        softmax512_kernel<<<BATCH, 512, 0, stream>>>(dist, dist);
        fillzero_kernel<<<(BATCH * DM) / 256, 256, 0, stream>>>(agg, BATCH * DM);
        if (hop == 0)
            gemm_hop_kernel<float><<<dim3(DM / 128, NROWS / 128), 256,
                                     GEMM_SMEM_BYTES, stream>>>(
                state, Wh, bias, dist, sNext, agg);
        else
            gemm_hop_kernel<_Float16><<<dim3(DM / 128, NROWS / 128), 256,
                                        GEMM_SMEM_BYTES, stream>>>(
                sPrev, Wh + (size_t)hop * DM * DM, bias + (size_t)hop * DM,
                dist, sNext, agg);
        float* qOutP = (hop == 2) ? outQ : (hop == 0 ? qA : qB);
        ln512_kernel<<<BATCH, 512, 0, stream>>>(qCur, agg, gamma, beta, qOutP, qn);
        qCur  = qOutP;
        sPrev = sNext;
        sNext = (sNext == s1) ? s2 : s1;
    }
    // final_dist = softmax(cos(q_final, s_final) * TEMP)
    dot_cos_kernel<_Float16><<<NROWS / 8, 256, 0, stream>>>(qCur, sPrev, qn, dist);
    softmax512_kernel<<<BATCH, 512, 0, stream>>>(dist, outDist);
}
